// RouterConv_1391569404261
// MI455X (gfx1250) — matrix operand order repaired and checked by reading, not compiled, not run on hardware
//
#include <hip/hip_runtime.h>
#include <hip/hip_bf16.h>
#include <math.h>

typedef __attribute__((ext_vector_type(8)))  float  v8f;
typedef __attribute__((ext_vector_type(8)))  __bf16 v8bf;
typedef __attribute__((ext_vector_type(16))) __bf16 v16bf;

#define Bn      64
#define C_IN    256
#define C_OUT   256
#define HW      256      // 16x16 pixels
#define KEXP    8
#define RED_CH  16
#define KSQ     9        // 3x3 taps
#define KSZ     2304     // C_IN * KSQ  (GEMM reduction length)

// ---------------- Kernel 1: routing[b, c, k] ----------------
__global__ __launch_bounds__(256) void routing_kernel(
    const float* __restrict__ x, const float* __restrict__ w1,
    const float* __restrict__ w2, const float* __restrict__ w_exp,
    float* __restrict__ routing)
{
  __shared__ float gapS[C_IN];
  __shared__ float poolS[16];
  __shared__ float hidS[RED_CH];
  __shared__ float ewS[KEXP];
  const int c = threadIdx.x;      // channel
  const int b = blockIdx.x;
  if (c < 16) poolS[c] = 0.f;
  __syncthreads();

  // per-channel sum + per-4x4-cell partial sums (fully unrolled -> no spills)
  const float4* xp = (const float4*)(x + ((size_t)b * C_IN + c) * HW);
  float s = 0.f;
  float ps[16];
  #pragma unroll
  for (int j = 0; j < 16; ++j) ps[j] = 0.f;
  #pragma unroll
  for (int y = 0; y < 16; ++y) {
    #pragma unroll
    for (int m = 0; m < 4; ++m) {
      float4 v = xp[y * 4 + m];
      float t4 = v.x + v.y + v.z + v.w;
      s += t4;
      ps[(y >> 2) * 4 + m] += t4;
    }
  }
  gapS[c] = s * (1.f / 256.f);
  #pragma unroll
  for (int j = 0; j < 16; ++j) atomicAdd(&poolS[j], ps[j]);  // ds_add_f32
  __syncthreads();

  if (c < RED_CH) {                       // hidden = relu(gap @ w1^T)
    float h = 0.f;
    for (int i = 0; i < C_IN; ++i) h += gapS[i] * w1[c * C_IN + i];
    hidS[c] = fmaxf(h, 0.f);
  }
  if (c < KEXP) {                         // ew = cp @ w_exp^T
    float e = 0.f;
    const float inv = 1.f / (16.f * 256.f);
    #pragma unroll
    for (int p = 0; p < 16; ++p) e += poolS[p] * inv * w_exp[c * 16 + p];
    ewS[c] = e;
  }
  __syncthreads();

  float cw = 0.f;                         // cw = hidden @ w2^T
  #pragma unroll
  for (int j = 0; j < RED_CH; ++j) cw += hidS[j] * w2[c * RED_CH + j];
  float* rp = routing + ((size_t)b * C_IN + c) * KEXP;
  #pragma unroll
  for (int k = 0; k < KEXP; ++k) {
    float z = cw * ewS[k];
    rp[k] = 1.f / (1.f + __expf(-z));     // sigmoid
  }
}

// ---------------- Kernel 2: fused combine + implicit-GEMM conv ----------------
// LDS layout (dynamic):
//   routS : 2048 f32                                            =   8192 B
//   combS : [16][2312] bf16 (reduction dim = uv*256+i, padded)  =  73984 B
//   xTS   : [18][18][264] bf16 (padded image, channel-innermost)= 171072 B
#define CPITCH   2312
#define XPITCH   264
#define XT_HALFS (18 * 18 * XPITCH)          // 85536
#define LDS_COMB 8192
#define LDS_XT   (LDS_COMB + 16 * CPITCH * 2)        // 82176
#define LDS_TOTAL (LDS_XT + XT_HALFS * 2)            // 253248

__global__ __launch_bounds__(256) void rconv_wmma_kernel(
    const float* __restrict__ x, const float* __restrict__ weight,
    const float* __restrict__ routing, float* __restrict__ out)
{
  extern __shared__ char smem[];
  float*  routS = (float*)(smem);
  __bf16* combS = (__bf16*)(smem + LDS_COMB);
  __bf16* xTS   = (__bf16*)(smem + LDS_XT);

  const int t  = threadIdx.x;
  const int o0 = blockIdx.x * 16;     // output-channel tile
  const int b  = blockIdx.y;          // batch sample

  // per-thread reduction indices: lane-contiguous r = t + j*256 (coalesced weight reads)
  int idx_i[KSQ], idx_uv[KSQ];
  #pragma unroll
  for (int j = 0; j < KSQ; ++j) {
    int r = t + j * 256;
    idx_i[j]  = r / KSQ;
    idx_uv[j] = r % KSQ;
  }

  // phase 0: routing[b] -> LDS (2048 floats)
  const float* rsrc = routing + (size_t)b * (C_IN * KEXP);
  #pragma unroll
  for (int i = 0; i < 8; ++i) routS[t + i * 256] = rsrc[t + i * 256];

  // phase 1a: zero padded xT (border = conv padding)
  {
    uint4 z = make_uint4(0u, 0u, 0u, 0u);
    uint4* p = (uint4*)xTS;
    for (int idx = t; idx < XT_HALFS / 8; idx += 256) p[idx] = z;
  }
  __syncthreads();

  // hoist routing factors for this thread's 9 reduction slots (LDS -> VGPRs, once)
  float rv[KSQ][KEXP];
  #pragma unroll
  for (int j = 0; j < KSQ; ++j) {
    #pragma unroll
    for (int k = 0; k < KEXP; ++k) rv[j][k] = routS[idx_i[j] * KEXP + k];
  }

  // phase 1b: x[b] (f32, NCHW) -> xT[y+1][x+1][i] (bf16), coalesced b128 reads
  {
    const float4* xp = (const float4*)(x + (size_t)b * (C_IN * HW));
    for (int q = t; q < C_IN * HW / 4; q += 256) {
      float4 v = xp[q];
      int i  = q >> 6;
      int y  = (q >> 2) & 15;
      int x4 = (q & 3) * 4;
      int base = ((y + 1) * 18 + (x4 + 1)) * XPITCH + i;
      xTS[base]              = (__bf16)v.x;
      xTS[base + XPITCH]     = (__bf16)v.y;
      xTS[base + 2 * XPITCH] = (__bf16)v.z;
      xTS[base + 3 * XPITCH] = (__bf16)v.w;
    }
  }

  // phase 2: combS[o][uv*256 + i] = sum_k rout[i][k] * weight[k, o0+o, r]
  // weight reads are unit-stride across lanes (r = t + j*256) -> fully coalesced.
  #pragma unroll 1
  for (int o = 0; o < 16; ++o) {
    const float* wo = weight + (size_t)(o0 + o) * KSZ + t;
    float acc[KSQ];
    #pragma unroll
    for (int j = 0; j < KSQ; ++j) acc[j] = 0.f;
    #pragma unroll
    for (int k = 0; k < KEXP; ++k) {
      const float* wk = wo + (size_t)k * (C_OUT * KSZ);
      #pragma unroll
      for (int j = 0; j < KSQ; ++j) acc[j] += rv[j][k] * wk[j * 256];
    }
    #pragma unroll
    for (int j = 0; j < KSQ; ++j)
      combS[o * CPITCH + idx_uv[j] * 256 + idx_i[j]] = (__bf16)acc[j];
  }
  __syncthreads();

  // phase 3: implicit GEMM, D[16 o x 16 px] += A[16 x 32] * B[32 x 16], 72 K-chunks
  const int wave = t >> 5;
  const int lane = t & 31;
  const int mn   = lane & 15;   // A: row m; B/D: column n
  const int hi   = lane >> 4;   // half-wave selector (ISA 7.12.2 layouts)
  const int ry0  = wave * 2;    // each wave: two 16-pixel image rows
  const int ry1  = wave * 2 + 1;

  v8f acc0 = {};
  v8f acc1 = {};

  #pragma unroll
  for (int uv = 0; uv < KSQ; ++uv) {
    const int u = uv / 3, v = uv % 3;
    #pragma unroll
    for (int kc = 0; kc < 8; ++kc) {
      // A fragment: lane holds row m, 16 consecutive K (hi*16 .. +15)
      int abase = mn * CPITCH + uv * 256 + kc * 32 + hi * 16;
      v8bf alo = *(const v8bf*)(combS + abase);
      v8bf ahi = *(const v8bf*)(combS + abase + 8);
      v16bf A = __builtin_shufflevector(alo, ahi,
                  0, 1, 2, 3, 4, 5, 6, 7, 8, 9, 10, 11, 12, 13, 14, 15);

      // B fragment: lane holds column n, 16 consecutive K (= channels) -> contiguous LDS
      int ibase = kc * 32 + hi * 16;
      int bidx0 = ((ry0 + u) * 18 + (mn + v)) * XPITCH + ibase;
      v8bf blo0 = *(const v8bf*)(xTS + bidx0);
      v8bf bhi0 = *(const v8bf*)(xTS + bidx0 + 8);
      v16bf B0 = __builtin_shufflevector(blo0, bhi0,
                  0, 1, 2, 3, 4, 5, 6, 7, 8, 9, 10, 11, 12, 13, 14, 15);
      acc0 = __builtin_amdgcn_wmma_f32_16x16x32_bf16(
                 false, A, false, B0, (short)0, acc0, false, false);

      int bidx1 = ((ry1 + u) * 18 + (mn + v)) * XPITCH + ibase;
      v8bf blo1 = *(const v8bf*)(xTS + bidx1);
      v8bf bhi1 = *(const v8bf*)(xTS + bidx1 + 8);
      v16bf B1 = __builtin_shufflevector(blo1, bhi1,
                  0, 1, 2, 3, 4, 5, 6, 7, 8, 9, 10, 11, 12, 13, 14, 15);
      acc1 = __builtin_amdgcn_wmma_f32_16x16x32_bf16(
                 false, A, false, B1, (short)0, acc1, false, false);
    }
  }

  // store: C/D layout -> VGPR j: M = hi*8 + j, N = mn ; coalesced 64B per M-row
  float* obase = out + ((size_t)b * C_OUT + o0) * HW;
  #pragma unroll
  for (int j = 0; j < 8; ++j) {
    int M = hi * 8 + j;
    obase[(size_t)M * HW + ry0 * 16 + mn] = acc0[j];
    obase[(size_t)M * HW + ry1 * 16 + mn] = acc1[j];
  }
}

extern "C" void kernel_launch(void* const* d_in, const int* in_sizes, int n_in,
                              void* d_out, int out_size, void* d_ws, size_t ws_size,
                              hipStream_t stream) {
  const float* x      = (const float*)d_in[0];
  const float* w1     = (const float*)d_in[1];
  const float* w2     = (const float*)d_in[2];
  const float* w_exp  = (const float*)d_in[3];
  const float* weight = (const float*)d_in[4];
  float* out = (float*)d_out;
  float* routing = (float*)d_ws;   // 64*256*8 f32 = 512 KB

  routing_kernel<<<Bn, 256, 0, stream>>>(x, w1, w2, w_exp, routing);

  dim3 grid(C_OUT / 16, Bn);       // (o-tile, batch)
  rconv_wmma_kernel<<<grid, 256, LDS_TOTAL, stream>>>(x, weight, routing, out);
}
